// NS_25031069401540
// MI455X (gfx1250) — matrix operand order repaired and checked by reading, not compiled, not run on hardware
//
#include <hip/hip_runtime.h>
#include <math.h>

#define TPD 46
#define TRD 24
#define NT0 40
#define NCH 20
#define NFQ (64*64*24)   /* 98304  frequency points */
#define NPPD (64*64*46)  /* 188416 padded spatial points */
#define NPSD (64*64*40)  /* 163840 unpadded spatial points */

typedef __attribute__((ext_vector_type(16))) _Float16 v16h;
typedef __attribute__((ext_vector_type(8)))  float    v8f;

__device__ inline float gelu_f(float x) {
    return 0.5f * x * (1.0f + erff(x * 0.70710678118654752f));
}

// ---------------------------------------------------------------------------
// v_wmma_f32_16x16x32_f16 fragment layouts (wave32, CDNA5 ISA):
//  A (16x32): lane row m = lane&15; halves 0..7 -> K kb+0..7, 8..15 -> K kb+16..23,
//             kb = (lane>=16 ? 8 : 0)
//  B (32x16): lane col n = lane&15; halves 0..7 -> K kb+0..7, 8..15 -> K kb+16..23, kb = (lane>=16?8:0)
//  C/D: elem r -> row r + 8*(lane>=16), col lane&15
// Addressing: row m -> (m/M2)*sM1 + (m%M2)*sM2 ; contraction k -> k*sK.
// OOB rows/cols are CLAMPED (loads always legal, values garbage but never
// stored); only the K-remainder slab zero-selects (exact zeros required).
// B matrices are small (K*N <= 4096) and staged whole into LDS via
// global_load_async_to_lds_b32 (ASYNCcnt) once per workgroup.
// ---------------------------------------------------------------------------
struct AAddr { long ro; int kb; };

__device__ inline AAddr a_addr(int m0, int M, int M2, long sM1, long sM2) {
    int lane = threadIdx.x & 31;
    int m = m0 + (lane & 15);
    if (m >= M) m = M - 1;
    AAddr s;
    s.ro = (long)(m / M2) * sM1 + (long)(m % M2) * sM2;
    s.kb = (lane >> 4) << 3;
    return s;
}

__device__ inline v16h a_load_fast(const float* __restrict__ A, AAddr s, int kk, long sK) {
    v16h r;
#pragma unroll
    for (int e = 0; e < 8; ++e) {
        r[e]     = (_Float16)A[s.ro + (long)(kk + s.kb + e) * sK];
        r[8 + e] = (_Float16)A[s.ro + (long)(kk + s.kb + 16 + e) * sK];
    }
    return r;
}
__device__ inline v16h a_load_edge(const float* __restrict__ A, AAddr s, int kk, int K, long sK) {
    v16h r;
#pragma unroll
    for (int e = 0; e < 8; ++e) {
        int k1 = kk + s.kb + e, k2 = kk + s.kb + 16 + e;
        float v1 = A[s.ro + (long)((k1 < K) ? k1 : 0) * sK];
        float v2 = A[s.ro + (long)((k2 < K) ? k2 : 0) * sK];
        r[e]     = (_Float16)((k1 < K) ? v1 : 0.0f);
        r[8 + e] = (_Float16)((k2 < K) ? v2 : 0.0f);
    }
    return r;
}
__device__ inline v16h neg_frag(v16h a) {
    v16h r;
#pragma unroll
    for (int e = 0; e < 16; ++e) r[e] = (_Float16)(-(float)a[e]);
    return r;
}

// ---- async global -> LDS staging of a whole [K x N] B matrix (row-major) ----
__device__ inline void stage_b_async(const float* __restrict__ G, long sK, long sN,
                                     int K, int N, float* S) {
    int total = K * N;
    for (int idx = threadIdx.x; idx < total; idx += blockDim.x) {
        int k = idx / N, n = idx % N;
        unsigned long long ga =
            (unsigned long long)(const char*)(G + (long)k * sK + (long)n * sN);
        unsigned la = (unsigned)(unsigned long long)(const char*)(S + idx);
        asm volatile("global_load_async_to_lds_b32 %0, %1, off"
                     :: "v"(la), "v"(ga) : "memory");
    }
    asm volatile("s_wait_asynccnt 0x0" ::: "memory");
    __syncthreads();
}

// ---- B fragment loads from LDS (contiguous, ds_load) ----
__device__ inline int b_col(int n0, int N) {
    int n = n0 + (threadIdx.x & 15);
    return (n < N) ? n : N - 1;       // clamped; garbage cols never stored
}
__device__ inline v16h b_lds_fast(const float* S, int nc, int kk, int N) {
    int kb = kk + (((threadIdx.x & 31) >> 4) << 3);
    v16h r;
#pragma unroll
    for (int e = 0; e < 16; ++e)
        r[e] = (_Float16)S[(kb + e + ((e >> 3) << 3)) * N + nc];
    return r;
}
__device__ inline v16h b_lds_edge(const float* S, int nc, int kk, int K, int N) {
    int kb = kk + (((threadIdx.x & 31) >> 4) << 3);
    v16h r;
#pragma unroll
    for (int e = 0; e < 16; ++e) {
        int k = kb + e + ((e >> 3) << 3);
        float v = S[((k < K) ? k : 0) * N + nc];
        r[e] = (_Float16)((k < K) ? v : 0.0f);
    }
    return r;
}

__device__ inline void store_c16(float* __restrict__ C, const v8f acc, int m0, int n0,
                                 int M, int N, int M2, long sM1, long sM2, long sN,
                                 const float* __restrict__ bias, int act) {
    int lane = threadIdx.x & 31;
    int n = n0 + (lane & 15);
    int rbase = m0 + ((lane >> 4) << 3);
    if (m0 + 16 <= M && n0 + 16 <= N) {         // full tile: unguarded
        float bv = bias ? bias[n] : 0.0f;
        int q = rbase / M2, rm = rbase % M2;
#pragma unroll
        for (int r = 0; r < 8; ++r) {
            float v = acc[r] + bv;
            if (act) v = gelu_f(v);
            C[(long)q * sM1 + (long)rm * sM2 + (long)n * sN] = v;
            if (++rm == M2) { rm = 0; ++q; }
        }
    } else {
        if (n >= N) return;
        float bv = bias ? bias[n] : 0.0f;
        int q = rbase / M2, rm = rbase % M2;
#pragma unroll
        for (int r = 0; r < 8; ++r) {
            if (rbase + r < M) {
                float v = acc[r] + bv;
                if (act) v = gelu_f(v);
                C[(long)q * sM1 + (long)rm * sM2 + (long)n * sN] = v;
            }
            if (++rm == M2) { rm = 0; ++q; }
        }
    }
}

#define WMMA_F16(A_, B_, C_) \
    __builtin_amdgcn_wmma_f32_16x16x32_f16(false, (A_), false, (B_), (short)0, (C_), false, false)

// ------------------- real GEMM: 32x32 macro-tile per wave -------------------
__global__ __launch_bounds__(128) void gemm_wmma(
    const float* __restrict__ A, const float* __restrict__ B,
    float* __restrict__ C, const float* __restrict__ bias,
    int M, int N, int K, int M2,
    long sAm1, long sAm2, long sAk,
    long sBk, long sBn,
    long sCm1, long sCm2, long sCn, int act) {
    __shared__ float sB[4096];                  // whole B, [K][N] row-major
    stage_b_async(B, sBk, sBn, K, N, sB);

    int wave = blockIdx.x * (blockDim.x >> 5) + (threadIdx.x >> 5);
    int nwaves = gridDim.x * (blockDim.x >> 5);
    int MT = (M + 31) >> 5, NT = (N + 31) >> 5;
    long tiles = (long)MT * NT;
    int kmain = K & ~31;
    for (long t = wave; t < tiles; t += nwaves) {
        int m0 = (int)(t / NT) << 5;
        int n0 = (int)(t % NT) << 5;
        AAddr a0 = a_addr(m0, M, M2, sAm1, sAm2);
        AAddr a1 = a_addr(m0 + 16, M, M2, sAm1, sAm2);
        int nc0 = b_col(n0, N), nc1 = b_col(n0 + 16, N);
        v8f c00 = {}, c01 = {}, c10 = {}, c11 = {};
        for (int kk = 0; kk < kmain; kk += 32) {
            v16h va0 = a_load_fast(A, a0, kk, sAk);
            v16h va1 = a_load_fast(A, a1, kk, sAk);
            v16h vb0 = b_lds_fast(sB, nc0, kk, N);
            v16h vb1 = b_lds_fast(sB, nc1, kk, N);
            c00 = WMMA_F16(va0, vb0, c00);
            c01 = WMMA_F16(va0, vb1, c01);
            c10 = WMMA_F16(va1, vb0, c10);
            c11 = WMMA_F16(va1, vb1, c11);
        }
        if (kmain < K) {
            v16h va0 = a_load_edge(A, a0, kmain, K, sAk);
            v16h va1 = a_load_edge(A, a1, kmain, K, sAk);
            v16h vb0 = b_lds_edge(sB, nc0, kmain, K, N);
            v16h vb1 = b_lds_edge(sB, nc1, kmain, K, N);
            c00 = WMMA_F16(va0, vb0, c00);
            c01 = WMMA_F16(va0, vb1, c01);
            c10 = WMMA_F16(va1, vb0, c10);
            c11 = WMMA_F16(va1, vb1, c11);
        }
        store_c16(C, c00, m0,      n0,      M, N, M2, sCm1, sCm2, sCn, bias, act);
        store_c16(C, c01, m0,      n0 + 16, M, N, M2, sCm1, sCm2, sCn, bias, act);
        store_c16(C, c10, m0 + 16, n0,      M, N, M2, sCm1, sCm2, sCn, bias, act);
        store_c16(C, c11, m0 + 16, n0 + 16, M, N, M2, sCm1, sCm2, sCn, bias, act);
    }
}

// ----------------- complex GEMM: 16x32 macro-tile per wave ------------------
// C = A * B (complex). Ai==null -> real input. Ci==null -> real output.
__global__ __launch_bounds__(128) void cgemm_wmma(
    const float* __restrict__ Ar, const float* __restrict__ Ai,
    const float* __restrict__ Br, const float* __restrict__ Bi,
    float* __restrict__ Cr, float* __restrict__ Ci,
    int M, int N, int K, int M2,
    long sAm1, long sAm2, long sAk,
    long sBk, long sBn,
    long sCm1, long sCm2, long sCn) {
    __shared__ float sBr[4096];
    __shared__ float sBi[4096];
    stage_b_async(Br, sBk, sBn, K, N, sBr);
    stage_b_async(Bi, sBk, sBn, K, N, sBi);

    int wave = blockIdx.x * (blockDim.x >> 5) + (threadIdx.x >> 5);
    int nwaves = gridDim.x * (blockDim.x >> 5);
    int MT = (M + 15) >> 4, NT = (N + 31) >> 5;
    long tiles = (long)MT * NT;
    int kmain = K & ~31;
    for (long t = wave; t < tiles; t += nwaves) {
        int m0 = (int)(t / NT) << 4;
        int n0 = (int)(t % NT) << 5;
        AAddr aa = a_addr(m0, M, M2, sAm1, sAm2);
        int nc0 = b_col(n0, N), nc1 = b_col(n0 + 16, N);
        v8f r0 = {}, i0 = {}, r1 = {}, i1 = {};
        for (int kk = 0; kk <= kmain; kk += 32) {
            bool edge = (kk == kmain);
            if (edge && kmain >= K) break;
            v16h ar  = edge ? a_load_edge(Ar, aa, kk, K, sAk) : a_load_fast(Ar, aa, kk, sAk);
            v16h br0 = edge ? b_lds_edge(sBr, nc0, kk, K, N) : b_lds_fast(sBr, nc0, kk, N);
            v16h bi0 = edge ? b_lds_edge(sBi, nc0, kk, K, N) : b_lds_fast(sBi, nc0, kk, N);
            v16h br1 = edge ? b_lds_edge(sBr, nc1, kk, K, N) : b_lds_fast(sBr, nc1, kk, N);
            v16h bi1 = edge ? b_lds_edge(sBi, nc1, kk, K, N) : b_lds_fast(sBi, nc1, kk, N);
            r0 = WMMA_F16(ar, br0, r0);
            r1 = WMMA_F16(ar, br1, r1);
            if (Ci) { i0 = WMMA_F16(ar, bi0, i0); i1 = WMMA_F16(ar, bi1, i1); }
            if (Ai) {
                v16h ai  = edge ? a_load_edge(Ai, aa, kk, K, sAk) : a_load_fast(Ai, aa, kk, sAk);
                v16h ain = neg_frag(ai);
                r0 = WMMA_F16(ain, bi0, r0);
                r1 = WMMA_F16(ain, bi1, r1);
                if (Ci) { i0 = WMMA_F16(ai, br0, i0); i1 = WMMA_F16(ai, br1, i1); }
            }
        }
        store_c16(Cr, r0, m0, n0,      M, N, M2, sCm1, sCm2, sCn, nullptr, 0);
        store_c16(Cr, r1, m0, n0 + 16, M, N, M2, sCm1, sCm2, sCn, nullptr, 0);
        if (Ci) {
            store_c16(Ci, i0, m0, n0,      M, N, M2, sCm1, sCm2, sCn, nullptr, 0);
            store_c16(Ci, i1, m0, n0 + 16, M, N, M2, sCm1, sCm2, sCn, nullptr, 0);
        }
    }
}

// ------------------------- table / grid init -------------------------------
__global__ void init_tables_kernel(float* TFre, float* TFim, float* TIre, float* TIim,
                                   float* FFre, float* FFim, float* FIre, float* FIim,
                                   float* gridf, float* grids) {
    const float PI2 = 6.283185307179586f;
    const int S1 = TPD * TRD, S2 = TRD * TPD, S3 = 64 * 64, S4 = 64 * 64;
    int total = S1 + S2 + S3 + S4 + NFQ + NPPD;
    for (int id = blockIdx.x * blockDim.x + threadIdx.x; id < total;
         id += gridDim.x * blockDim.x) {
        int e = id;
        if (e < S1) {                       // forward rfft(T): [t][k], 46x24
            int t = e / TRD, k = e % TRD;
            float ang = PI2 * (float)(t * k) / (float)TPD;
            TFre[e] = cosf(ang);
            TFim[e] = -sinf(ang);
            continue;
        }
        e -= S1;
        if (e < S2) {                       // inverse irfft(T): [k][t], 24x46
            int k = e / TPD, t = e % TPD;
            float ang = PI2 * (float)(k * t) / (float)TPD;
            float cre = (k == 0 || k == TPD / 2) ? 1.0f : 2.0f;
            float cim = (k == 0 || k == TPD / 2) ? 0.0f : 2.0f;
            TIre[e] = (cre / (float)TPD) * cosf(ang);
            TIim[e] = (cim / (float)TPD) * sinf(ang);   // Re = a*Br - b*Bi
            continue;
        }
        e -= S2;
        if (e < S3) {                       // forward 64-pt fft: [j][k]
            int j = e / 64, k = e % 64;
            float ang = PI2 * (float)(j * k) / 64.0f;
            FFre[e] = cosf(ang);
            FFim[e] = -sinf(ang);
            continue;
        }
        e -= S3;
        if (e < S4) {                       // inverse 64-pt fft: [k][m], 1/64
            int k = e / 64, m = e % 64;
            float ang = PI2 * (float)(k * m) / 64.0f;
            FIre[e] = cosf(ang) / 64.0f;
            FIim[e] = sinf(ang) / 64.0f;
            continue;
        }
        e -= S4;
        if (e < NFQ) {                      // frequency grid (fx, fy, ft)
            int x = e / (64 * TRD), y = (e / TRD) % 64, t = e % TRD;
            float fx = (x < 32) ? (float)x / 64.0f : (float)(x - 64) / 64.0f;
            float fy = (y < 32) ? (float)y / 64.0f : (float)(y - 64) / 64.0f;
            gridf[3 * e + 0] = fx; gridf[3 * e + 1] = fy;
            gridf[3 * e + 2] = (float)t / (float)TPD;
            continue;
        }
        e -= NFQ;
        {                                   // spatial grid over padded T
            int x = e / (64 * TPD), y = (e / TPD) % 64, t = e % TPD;
            grids[3 * e + 0] = (float)x / 63.0f;
            grids[3 * e + 1] = (float)y / 63.0f;
            grids[3 * e + 2] = (float)t / 45.0f;
        }
    }
}

// ------------------------------- fc0 ---------------------------------------
__global__ void fc0_kernel(const float* __restrict__ X, const float* __restrict__ W,
                           const float* __restrict__ B, float* __restrict__ H) {
    for (int id = blockIdx.x * blockDim.x + threadIdx.x; id < NPPD;
         id += gridDim.x * blockDim.x) {
        int x = id / (64 * TPD), y = (id / TPD) % 64, t = id % TPD;
        if (t >= NT0) {
            for (int c = 0; c < NCH; ++c) H[(long)c * NPPD + id] = 0.0f;
            continue;
        }
        const float* xin = X + ((long)((x * 64 + y) * NT0 + t)) * 10;
        float gx = (float)x / 63.0f, gy = (float)y / 63.0f, gt = (float)t / 39.0f;
        for (int c = 0; c < NCH; ++c) {
            float acc = B[c];
            for (int d = 0; d < 10; ++d) acc += xin[d] * W[d * NCH + c];
            acc += gx * W[10 * NCH + c] + gy * W[11 * NCH + c] + gt * W[12 * NCH + c];
            H[(long)c * NPPD + id] = acc;
        }
    }
}

// --------------------- spectral multiply (per o-chunk) ---------------------
__global__ void spectral_mul_kernel(const float* __restrict__ xfr, const float* __restrict__ xfi,
                                    const float* __restrict__ wr, const float* __restrict__ wi,
                                    float* __restrict__ outr, float* __restrict__ outi) {
    int n = NCH * NFQ;
    for (int id = blockIdx.x * blockDim.x + threadIdx.x; id < n;
         id += gridDim.x * blockDim.x) {
        int i = id / NFQ, p = id % NFQ;
        float xr = xfr[id], xi = xfi[id];
        float a = wr[(long)p * NCH + i], b = wi[(long)p * NCH + i];
        outr[id] = xr * a - xi * b;
        outi[id] = xr * b + xi * a;
    }
}

// ----------- i-contraction + pointwise channel mix + bias + gelu -----------
__global__ void combine_kernel(const float* __restrict__ xs, const float* __restrict__ wout,
                               const float* __restrict__ h, const float* __restrict__ Wl,
                               const float* __restrict__ bl, float* __restrict__ hn,
                               int o, int act) {
    for (int p = blockIdx.x * blockDim.x + threadIdx.x; p < NPPD;
         p += gridDim.x * blockDim.x) {
        float acc = bl[o];
        for (int i = 0; i < NCH; ++i) {
            acc += xs[(long)i * NPPD + p] * wout[(long)p * NCH + i];
            acc += h[(long)i * NPPD + p] * Wl[o * NCH + i];
        }
        if (act) acc = gelu_f(acc);
        hn[(long)o * NPPD + p] = acc;
    }
}

// -------------------- gather to point-major for fc1 ------------------------
__global__ void gather_fc1_kernel(const float* __restrict__ H, float* __restrict__ A) {
    for (int p = blockIdx.x * blockDim.x + threadIdx.x; p < NPSD;
         p += gridDim.x * blockDim.x) {
        int x = p / (64 * NT0), y = (p / NT0) % 64, t = p % NT0;
        long q = (long)((x * 64 + y) * TPD + t);
        for (int c = 0; c < NCH; ++c) A[(long)p * NCH + c] = H[(long)c * NPPD + q];
    }
}

// ------------------------------- fc2 ---------------------------------------
__global__ void fc2_kernel(const float* __restrict__ A, const float* __restrict__ W,
                           const float* __restrict__ B, float* __restrict__ OUT) {
    for (int p = blockIdx.x * blockDim.x + threadIdx.x; p < NPSD;
         p += gridDim.x * blockDim.x) {
        float acc = B[0];
        const float* row = A + (long)p * 128;
        for (int c = 0; c < 128; ++c) acc += row[c] * W[c];
        OUT[p] = acc;
    }
}

// ===========================================================================
extern "C" void kernel_launch(void* const* d_in, const int* in_sizes, int n_in,
                              void* d_out, int out_size, void* d_ws, size_t ws_size,
                              hipStream_t stream) {
    (void)in_sizes; (void)n_in; (void)out_size; (void)ws_size;
    const float* X    = (const float*)d_in[0];
    const float* fc0w = (const float*)d_in[1];
    const float* fc0b = (const float*)d_in[2];
    const float* fc1w = (const float*)d_in[3];
    const float* fc1b = (const float*)d_in[4];
    const float* fc2w = (const float*)d_in[5];
    const float* fc2b = (const float*)d_in[6];

    // ---- workspace layout (bump allocator) ----
    char* wsb = (char*)d_ws;
    size_t off = 0;
    auto alloc = [&](size_t nfloat) -> float* {
        float* p = (float*)(wsb + off);
        off += ((nfloat * sizeof(float) + 255) & ~(size_t)255);
        return p;
    };
    float* TFre = alloc(TPD * TRD); float* TFim = alloc(TPD * TRD);
    float* TIre = alloc(TRD * TPD); float* TIim = alloc(TRD * TPD);
    float* FFre = alloc(64 * 64);   float* FFim = alloc(64 * 64);
    float* FIre = alloc(64 * 64);   float* FIim = alloc(64 * 64);
    float* gridf = alloc((size_t)NFQ * 3);
    float* grids = alloc((size_t)NPPD * 3);
    float* hA    = alloc((size_t)NCH * NPPD);
    float* hB    = alloc((size_t)NCH * NPPD);
    float* xftre = alloc((size_t)NCH * NFQ);
    float* xftim = alloc((size_t)NCH * NFQ);
    float* cAre  = alloc((size_t)NCH * NFQ);
    float* cAim  = alloc((size_t)NCH * NFQ);
    float* cBre  = alloc((size_t)NCH * NFQ);
    float* cBim  = alloc((size_t)NCH * NFQ);
    float* fa3r  = alloc((size_t)NFQ * 32);
    float* fa3i  = alloc((size_t)NFQ * 32);
    // region reused by fc1 stage afterwards:
    float* tmp1  = alloc((size_t)NPPD * 32);
    float* tmp2  = alloc((size_t)NPPD * 32);
    float* sa3   = alloc((size_t)NPPD * 32);
    float* wr_o  = alloc((size_t)NFQ * NCH);
    float* wi_o  = alloc((size_t)NFQ * NCH);
    float* wout_o= alloc((size_t)NPPD * NCH);
    float* xs_o  = alloc((size_t)NCH * NPPD);
    float* fc1A   = tmp1;
    float* fc1out = (float*)((char*)tmp1 + (((size_t)NPSD * NCH * 4 + 255) & ~(size_t)255));

    // ---- launch helpers ----
    auto gemm = [&](const float* A, const float* B, float* C, const float* bias,
                    int M, int N, int K, int M2,
                    long sAm1, long sAm2, long sAk, long sBk, long sBn,
                    long sCm1, long sCm2, long sCn, int act) {
        long tiles = (long)((M + 31) / 32) * ((N + 31) / 32);
        int blocks = (int)((tiles + 3) / 4);
        if (blocks > 4096) blocks = 4096;
        if (blocks < 1) blocks = 1;
        gemm_wmma<<<blocks, 128, 0, stream>>>(A, B, C, bias, M, N, K, M2,
                                              sAm1, sAm2, sAk, sBk, sBn,
                                              sCm1, sCm2, sCn, act);
    };
    auto cgemm = [&](const float* Ar, const float* Ai, const float* Br, const float* Bi,
                     float* Cr, float* Ci, int M, int N, int K, int M2,
                     long sAm1, long sAm2, long sAk, long sBk, long sBn,
                     long sCm1, long sCm2, long sCn) {
        long tiles = (long)((M + 15) / 16) * ((N + 31) / 32);
        int blocks = (int)((tiles + 3) / 4);
        if (blocks > 4096) blocks = 4096;
        if (blocks < 1) blocks = 1;
        cgemm_wmma<<<blocks, 128, 0, stream>>>(Ar, Ai, Br, Bi, Cr, Ci, M, N, K, M2,
                                               sAm1, sAm2, sAk, sBk, sBn,
                                               sCm1, sCm2, sCn);
    };
    auto ew = [&](int n) {
        int b = (n + 255) / 256;
        if (b > 16384) b = 16384;
        return b;
    };
    auto run_net3 = [&](const float* gin, int NP, const float* const* P, float* a3) {
        gemm(gin, P[0], tmp1, P[1], NP, 32, 3,  1, 3,  0, 1, 32, 1, 32, 0, 1, 1);
        gemm(tmp1, P[2], tmp2, P[3], NP, 32, 32, 1, 32, 0, 1, 32, 1, 32, 0, 1, 1);
        gemm(tmp2, P[4], a3,  P[5], NP, 32, 32, 1, 32, 0, 1, 32, 1, 32, 0, 1, 1);
    };

    init_tables_kernel<<<1024, 256, 0, stream>>>(TFre, TFim, TIre, TIim,
                                                 FFre, FFim, FIre, FIim, gridf, grids);
    fc0_kernel<<<ew(NPPD), 256, 0, stream>>>(X, fc0w, fc0b, hA);

    float* hcur = hA;
    float* hnxt = hB;
    for (int L = 0; L < 4; ++L) {
        int base = 7 + L * 26;
        const float* pr[8]; const float* pim[8]; const float* po[8];
        for (int j = 0; j < 8; ++j) {
            pr[j]  = (const float*)d_in[base + j];
            pim[j] = (const float*)d_in[base + 8 + j];
            po[j]  = (const float*)d_in[base + 16 + j];
        }
        const float* Wl = (const float*)d_in[base + 24];
        const float* bl = (const float*)d_in[base + 25];

        // forward rfftn(h): T (real->cplx), then Y, then X  (DFT-as-GEMM)
        cgemm(hcur, nullptr, TFre, TFim, cAre, cAim,
              NCH * 64 * 64, TRD, TPD, 1, TPD, 0, 1, TRD, 1, TRD, 0, 1);
        cgemm(cAre, cAim, FFre, FFim, cBre, cBim,                     // Y axis
              NCH * 64 * TRD, 64, 64, TRD, 64 * TRD, 1, TRD, 64, 1, 64 * TRD, 1, TRD);
        cgemm(cBre, cBim, FFre, FFim, xftre, xftim,                   // X axis
              NCH * 64 * TRD, 64, 64, 64 * TRD, 64 * 64 * TRD, 1, 64 * TRD,
              64, 1, 64 * 64 * TRD, 1, 64 * TRD);

        // weight-field MLP trunks (shared across o-chunks)
        run_net3(gridf, NFQ,  pr,  fa3r);
        run_net3(gridf, NFQ,  pim, fa3i);
        run_net3(grids, NPPD, po,  sa3);

        for (int o = 0; o < NCH; ++o) {
            // sliced last MLP layer: 32 -> 20 columns [o*20, o*20+20)
            gemm(fa3r, pr[6]  + o * NCH, wr_o,   pr[7]  + o * NCH,
                 NFQ,  NCH, 32, 1, 32, 0, 1, 400, 1, NCH, 0, 1, 0);
            gemm(fa3i, pim[6] + o * NCH, wi_o,   pim[7] + o * NCH,
                 NFQ,  NCH, 32, 1, 32, 0, 1, 400, 1, NCH, 0, 1, 0);
            gemm(sa3,  po[6]  + o * NCH, wout_o, po[7]  + o * NCH,
                 NPPD, NCH, 32, 1, 32, 0, 1, 400, 1, NCH, 0, 1, 0);

            spectral_mul_kernel<<<ew(NCH * NFQ), 256, 0, stream>>>(
                xftre, xftim, wr_o, wi_o, cAre, cAim);

            // irfftn: inverse X, inverse Y, then irfft T (real output)
            cgemm(cAre, cAim, FIre, FIim, cBre, cBim,
                  NCH * 64 * TRD, 64, 64, 64 * TRD, 64 * 64 * TRD, 1, 64 * TRD,
                  64, 1, 64 * 64 * TRD, 1, 64 * TRD);
            cgemm(cBre, cBim, FIre, FIim, cAre, cAim,
                  NCH * 64 * TRD, 64, 64, TRD, 64 * TRD, 1, TRD, 64, 1, 64 * TRD, 1, TRD);
            cgemm(cAre, cAim, TIre, TIim, xs_o, nullptr,
                  NCH * 64 * 64, TPD, TRD, 1, TRD, 0, 1, TPD, 1, TPD, 0, 1);

            combine_kernel<<<ew(NPPD), 256, 0, stream>>>(
                xs_o, wout_o, hcur, Wl, bl, hnxt, o, (L < 3) ? 1 : 0);
        }
        float* t = hcur; hcur = hnxt; hnxt = t;
    }

    gather_fc1_kernel<<<ew(NPSD), 256, 0, stream>>>(hcur, fc1A);
    gemm(fc1A, fc1w, fc1out, fc1b, NPSD, 128, NCH,
         1, NCH, 0, 1, 128, 1, 128, 0, 1, 1);
    fc2_kernel<<<ew(NPSD), 256, 0, stream>>>(fc1out, fc2w, fc2b, (float*)d_out);
}
